// PrunedPartitionAttention_77996606096039
// MI455X (gfx1250) — matrix operand order repaired and checked by reading, not compiled, not run on hardware
//
#include <hip/hip_runtime.h>

typedef __attribute__((ext_vector_type(16))) _Float16 v16h;
typedef __attribute__((ext_vector_type(8)))  float    v8f;
typedef __attribute__((ext_vector_type(4)))  unsigned int v4u;

#define DEV __device__ __forceinline__

union F16x16 { v16h v; v4u u[2]; };

// A-fragment (16x32 f16, M=lane&15): two 16B chunks at k0+hf*8 and k0+16+hf*8
DEV v16h ldA(const _Float16* row, int k0, int hf) {
    F16x16 f;
    const _Float16* p = row + k0 + hf * 8;
    f.u[0] = *(const v4u*)(p);
    f.u[1] = *(const v4u*)(p + 16);
    return f.v;
}
// B-fragment (32x16 f16, N=lane&15): two 16B chunks at k0+hf*8 and k0+16+hf*8
DEV v16h ldB(const _Float16* col, int k0, int hf) {
    F16x16 f;
    const _Float16* p = col + k0 + hf * 8;
    f.u[0] = *(const v4u*)(p);
    f.u[1] = *(const v4u*)(p + 16);
    return f.v;
}
DEV v8f wmma_f16(v16h a, v16h b, v8f c) {
    return __builtin_amdgcn_wmma_f32_16x16x32_f16(false, a, false, b, (short)0, c, false, false);
}

// ---------------- weight f32 -> f16 transpose ----------------
__global__ void k_cvtT(const float* __restrict__ W, _Float16* __restrict__ WT, int K, int N) {
    int i = blockIdx.x * 256 + threadIdx.x;
    if (i >= K * N) return;
    int n = i % N, k = i / N;
    WT[(size_t)n * K + k] = (_Float16)W[i];
}

// ---------------- window partition + prob + patch embed ----------------
__global__ void __launch_bounds__(256)
k_embed(const float* __restrict__ x, const float* __restrict__ Wpc,
        const float* __restrict__ bpc, const _Float16* __restrict__ WpeT,
        const float* __restrict__ bpe, float* __restrict__ tok,
        float* __restrict__ prob) {
    __shared__ float    tf[49 * 96];
    __shared__ _Float16 th[64 * 96];
    __shared__ float    lg[49 * 2];
    const int tid = threadIdx.x;
    const int win = blockIdx.x;
    const int b = win >> 10, wi = win & 1023, wh = wi >> 5, ww = wi & 31;
    for (int idx = tid; idx < 49 * 96; idx += 256) {
        int t = idx / 96, c = idx % 96;
        int y = wh * 7 + t / 7, xx = ww * 7 + t % 7;
        float v = x[(((size_t)b * 96 + c) * 224 + y) * 224 + xx];
        tf[idx] = v;
        th[t * 96 + c] = (_Float16)v;
    }
    for (int idx = tid; idx < 15 * 96; idx += 256) th[49 * 96 + idx] = (_Float16)0.f;
    __syncthreads();
    if (tid < 98) {
        int t = tid >> 1, cls = tid & 1;
        float a = bpc[cls];
        for (int c = 0; c < 96; c++) a += tf[t * 96 + c] * Wpc[c * 2 + cls];
        lg[t * 2 + cls] = a;
    }
    __syncthreads();
    if (tid < 49) {
        float l0 = lg[tid * 2], l1 = lg[tid * 2 + 1];
        float m = fmaxf(l0, l1);
        float e0 = __expf(l0 - m), e1 = __expf(l1 - m);
        prob[(size_t)win * 49 + tid] = e1 / (e0 + e1);
    }
    const int wave = tid >> 5, lane = tid & 31, hf = lane >> 4, ll = lane & 15;
    for (int tile = wave; tile < 128; tile += 8) {
        int mt = tile & 3, nt = tile >> 2;
        int n = nt * 16 + ll;
        const _Float16* arow = th + (mt * 16 + ll) * 96;
        const _Float16* bcol = WpeT + (size_t)n * 96;
        v8f acc = {};
#pragma unroll
        for (int kt = 0; kt < 3; kt++)
            acc = wmma_f16(ldA(arow, kt * 32, hf), ldB(bcol, kt * 32, hf), acc);
        float bb = bpe[n];
        if (mt != 3) {                        // all 16 rows valid (m <= 47)
#pragma unroll
            for (int r = 0; r < 8; r++) {
                int m = mt * 16 + r + 8 * hf;
                tok[((size_t)win * 49 + m) * 512 + n] = acc[r] + bb;
            }
        } else if (hf == 0) {                 // only m == 48 is valid
            tok[((size_t)win * 49 + 48) * 512 + n] = acc[0] + bb;
        }
    }
}

// ---------------- LayerNorm: tok(f32) -> xn(f16, 64 rows, pads zeroed) ----------------
__global__ void __launch_bounds__(256)
k_ln(const float* __restrict__ tok, const float* __restrict__ g,
     const float* __restrict__ bta, _Float16* __restrict__ xn) {
    const int win = blockIdx.x;
    const int wave = threadIdx.x >> 5, lane = threadIdx.x & 31;
    for (int t = wave; t < 49; t += 8) {
        const float* row = tok + ((size_t)win * 49 + t) * 512;
        float vx[8], vy[8], s = 0.f, sq = 0.f;
#pragma unroll
        for (int i = 0; i < 8; i++) {
            float2 u = *(const float2*)(row + lane * 2 + i * 64);
            vx[i] = u.x; vy[i] = u.y;
            s += u.x + u.y; sq += u.x * u.x + u.y * u.y;
        }
#pragma unroll
        for (int m = 1; m < 32; m <<= 1) { s += __shfl_xor(s, m, 32); sq += __shfl_xor(sq, m, 32); }
        float mean = s * (1.f / 512.f);
        float rstd = rsqrtf(sq * (1.f / 512.f) - mean * mean + 1e-5f);
        _Float16* orow = xn + ((size_t)win * 64 + t) * 512;
#pragma unroll
        for (int i = 0; i < 8; i++) {
            int c = lane * 2 + i * 64;
            float2 gg = *(const float2*)(g + c);
            float2 bb = *(const float2*)(bta + c);
            union { _Float16 h[2]; unsigned u; } pk;
            pk.h[0] = (_Float16)((vx[i] - mean) * rstd * gg.x + bb.x);
            pk.h[1] = (_Float16)((vy[i] - mean) * rstd * gg.y + bb.y);
            *(unsigned*)(orow + c) = pk.u;    // packed 2 x f16 store
        }
    }
    for (int t = 49 + wave; t < 64; t += 8) {
        unsigned* orow = (unsigned*)(xn + ((size_t)win * 64 + t) * 512);
        for (int i = lane; i < 256; i += 32) orow[i] = 0u;
    }
}

// ---------------- per (window, head): qkv GEMM + pruned attention ----------------
__global__ void __launch_bounds__(128)
k_attn(const _Float16* __restrict__ xn, const _Float16* __restrict__ WqkvT,
       const float* __restrict__ prob, const float* __restrict__ rpb,
       const float* __restrict__ sita, const float* __restrict__ Wthr,
       _Float16* __restrict__ oatt) {
    __shared__ _Float16 qs[64 * 64], ks[64 * 64], vts[64 * 64];
    __shared__ _Float16 aph[4 * 16 * 64];
    __shared__ float thr_s[64], prob_s[64], wthr_s[64], rpb_s[169 * 8];
    const int tid = threadIdx.x;
    const int win = blockIdx.x >> 3, h = blockIdx.x & 7;
    const int wave = tid >> 5, lane = tid & 31, hf = lane >> 4, ll = lane & 15;

    for (int i = tid; i < 169 * 8; i += 128) rpb_s[i] = rpb[i];
    if (tid < 64) {
        wthr_s[tid] = Wthr[tid];
        prob_s[tid] = (tid < 49) ? prob[(size_t)win * 49 + tid] : 2.0f;
    }
    // qkv for this head: 48 tiles (4 mt x {q,k,v} x 4 nt), K=512
    for (int t = wave; t < 48; t += 4) {
        int mt = t & 3, j = t >> 2;
        int seg = j >> 2, ntl = j & 3;
        int c = seg * 512 + h * 64 + ntl * 16 + ll;
        const _Float16* arow = xn + ((size_t)win * 64 + mt * 16 + ll) * 512;
        const _Float16* bcol = WqkvT + (size_t)c * 512;
        __builtin_prefetch((const void*)(bcol + 16 * 512), 0, 1);  // next weight column block
        v8f acc = {};
#pragma unroll 4
        for (int kt = 0; kt < 16; kt++)
            acc = wmma_f16(ldA(arow, kt * 32, hf), ldB(bcol, kt * 32, hf), acc);
        int d = ntl * 16 + ll;
#pragma unroll
        for (int r = 0; r < 8; r++) {
            int m = mt * 16 + r + 8 * hf;
            _Float16 hv = (_Float16)acc[r];
            if (seg == 0)      qs[m * 64 + d] = hv;
            else if (seg == 1) ks[m * 64 + d] = hv;
            else               vts[d * 64 + m] = hv;   // v stored transposed [d][token]
        }
    }
    __syncthreads();
    // threshold logits sigmoid(q @ W_thr) * sigmoid(-2)
    const float sgate = 1.f / (1.f + __expf(2.0f));
    if (tid < 64) {
        float dot = 0.f;
        for (int d = 0; d < 64; d++) dot += (float)qs[tid * 64 + d] * wthr_s[d];
        thr_s[tid] = (1.f / (1.f + __expf(-dot))) * sgate;
    }
    __syncthreads();
    float st = sita[h];
    float factor = 1.f / (2.f * st * st + 1e-6f);
    const int mt = wave;                      // 4 waves <-> 4 row tiles
    v8f dots[4];
#pragma unroll
    for (int nt = 0; nt < 4; nt++) {
        v8f acc = {};
        const _Float16* arow = qs + (mt * 16 + ll) * 64;
        const _Float16* bcol = ks + (nt * 16 + ll) * 64;
#pragma unroll
        for (int kt = 0; kt < 2; kt++)
            acc = wmma_f16(ldA(arow, kt * 32, hf), ldB(bcol, kt * 32, hf), acc);
        dots[nt] = acc;
    }
#pragma unroll
    for (int r = 0; r < 8; r++) {
        int m = mt * 16 + r + 8 * hf;
        float val[4];
#pragma unroll
        for (int nt = 0; nt < 4; nt++) {
            int n = nt * 16 + ll;
            float v = dots[nt][r] * 0.125f;   // DHEAD^-0.5
            if (m < 49 && n < 49) {
                int dy = m / 7 - n / 7, dx = m % 7 - n % 7;
                int idx = (dy + 6) * 13 + (dx + 6);
                float fy = dy * (1.f / 7.f), fx = dx * (1.f / 7.f);
                v += rpb_s[idx * 8 + h] + 0.01f * __expf(-factor * (fy * fy + fx * fx));
            } else v = -3.0e38f;
            val[nt] = v;
        }
        float mx = fmaxf(fmaxf(val[0], val[1]), fmaxf(val[2], val[3]));
#pragma unroll
        for (int sh = 1; sh < 16; sh <<= 1) mx = fmaxf(mx, __shfl_xor(mx, sh, 32));
        float ex[4], sm = 0.f, mn = 3.0e38f;
#pragma unroll
        for (int nt = 0; nt < 4; nt++) {
            int n = nt * 16 + ll;
            if (n < 49) { ex[nt] = __expf(val[nt] - mx); mn = fminf(mn, val[nt]); }
            else ex[nt] = 0.f;
            sm += ex[nt];
        }
#pragma unroll
        for (int sh = 1; sh < 16; sh <<= 1) { sm += __shfl_xor(sm, sh, 32); mn = fminf(mn, __shfl_xor(mn, sh, 32)); }
        float inv = 1.f / sm;
        float amax = inv;                        // exp(0)/sum
        float amin = __expf(mn - mx) * inv;
        float t2 = amin + thr_s[m] * (amax - amin);
        bool rowgate = prob_s[m] < 0.9f;         // per-query-row prune gate
        float ap[4], deno = 0.f;
#pragma unroll
        for (int nt = 0; nt < 4; nt++) {
            float a = ex[nt] * inv;
            ap[nt] = (rowgate && (a > t2)) ? a : 0.f;
            deno += ap[nt];
        }
#pragma unroll
        for (int sh = 1; sh < 16; sh <<= 1) deno += __shfl_xor(deno, sh, 32);
        float rinv = 1.f / (deno + 1e-6f);
#pragma unroll
        for (int nt = 0; nt < 4; nt++)
            aph[(wave * 16 + r + 8 * hf) * 64 + nt * 16 + ll] = (_Float16)(ap[nt] * rinv);
    }
    // out = attn_p @ v   (K = 64 tokens)
#pragma unroll
    for (int nt = 0; nt < 4; nt++) {
        v8f acc = {};
        const _Float16* arow = aph + (wave * 16 + ll) * 64;
        const _Float16* bcol = vts + (nt * 16 + ll) * 64;
#pragma unroll
        for (int kt = 0; kt < 2; kt++)
            acc = wmma_f16(ldA(arow, kt * 32, hf), ldB(bcol, kt * 32, hf), acc);
#pragma unroll
        for (int r = 0; r < 8; r++) {
            int m = mt * 16 + r + 8 * hf;
            oatt[((size_t)win * 64 + m) * 512 + h * 64 + nt * 16 + ll] = (_Float16)acc[r];
        }
    }
}

// ---------------- out-projection + residual ----------------
__global__ void __launch_bounds__(256)
k_proj(const _Float16* __restrict__ A, const _Float16* __restrict__ BT,
       const float* __restrict__ bias, float* __restrict__ tok) {
    const int win = blockIdx.x;
    const int wave = threadIdx.x >> 5, lane = threadIdx.x & 31, hf = lane >> 4, ll = lane & 15;
    for (int tile = wave; tile < 128; tile += 8) {
        int mt = tile & 3, nt = tile >> 2;
        int n = nt * 16 + ll;
        const _Float16* arow = A + ((size_t)win * 64 + mt * 16 + ll) * 512;
        const _Float16* bcol = BT + (size_t)n * 512;
        __builtin_prefetch((const void*)(bcol + 32 * 512), 0, 1);  // next iteration's columns
        v8f acc = {};
#pragma unroll 4
        for (int kt = 0; kt < 16; kt++)
            acc = wmma_f16(ldA(arow, kt * 32, hf), ldB(bcol, kt * 32, hf), acc);
        float bb = bias[n];
        if (mt != 3) {
#pragma unroll
            for (int r = 0; r < 8; r++) {
                int m = mt * 16 + r + 8 * hf;
                size_t o = ((size_t)win * 49 + m) * 512 + n;
                tok[o] += acc[r] + bb;
            }
        } else if (hf == 0) {
            size_t o = ((size_t)win * 49 + 48) * 512 + n;
            tok[o] += acc[0] + bb;
        }
    }
}

// ---------------- fused MLP: relu(xn@W1+b1)@W2+b2 + residual ----------------
__global__ void __launch_bounds__(256)
k_ffn(const _Float16* __restrict__ xn, const _Float16* __restrict__ W1T,
      const float* __restrict__ b1, const _Float16* __restrict__ W2T,
      const float* __restrict__ b2, float* __restrict__ tok) {
    __shared__ _Float16 h1[64 * 1024];      // 128 KB: half of the hidden activation
    const int win = blockIdx.x;
    const int wave = threadIdx.x >> 5, lane = threadIdx.x & 31, hf = lane >> 4, ll = lane & 15;
    v8f acc[16];
    v8f z = {};
#pragma unroll
    for (int i = 0; i < 16; i++) acc[i] = z;
    for (int hv = 0; hv < 2; hv++) {
        __syncthreads();                    // WAR vs previous half's ff2 reads
        for (int tile = wave; tile < 256; tile += 8) {
            int mt = tile & 3, nt = tile >> 2;
            int gn = hv * 1024 + nt * 16 + ll;
            const _Float16* arow = xn + ((size_t)win * 64 + mt * 16 + ll) * 512;
            const _Float16* bcol = W1T + (size_t)gn * 512;
            __builtin_prefetch((const void*)(bcol + 32 * 512), 0, 1);
            v8f a = z;
#pragma unroll 4
            for (int kt = 0; kt < 16; kt++)
                a = wmma_f16(ldA(arow, kt * 32, hf), ldB(bcol, kt * 32, hf), a);
            float bb = b1[gn];
#pragma unroll
            for (int r = 0; r < 8; r++)
                h1[(mt * 16 + r + 8 * hf) * 1024 + nt * 16 + ll] = (_Float16)fmaxf(a[r] + bb, 0.f);
        }
        __syncthreads();
#pragma unroll
        for (int i = 0; i < 16; i++) {      // 16 persistent output tiles per wave
            int tile = wave * 16 + i;
            int mt = tile & 3, nt = tile >> 2;
            int n = nt * 16 + ll;
            const _Float16* arow = h1 + (mt * 16 + ll) * 1024;
            const _Float16* bcol = W2T + (size_t)n * 2048 + hv * 1024;
#pragma unroll 4
            for (int kt = 0; kt < 32; kt++)
                acc[i] = wmma_f16(ldA(arow, kt * 32, hf), ldB(bcol, kt * 32, hf), acc[i]);
        }
    }
#pragma unroll
    for (int i = 0; i < 16; i++) {
        int tile = wave * 16 + i;
        int mt = tile & 3, nt = tile >> 2;
        int n = nt * 16 + ll;
        float bb = b2[n];
        if (mt != 3) {
#pragma unroll
            for (int r = 0; r < 8; r++) {
                int m = mt * 16 + r + 8 * hf;
                size_t o = ((size_t)win * 49 + m) * 512 + n;
                tok[o] += acc[i][r] + bb;
            }
        } else if (hf == 0) {
            size_t o = ((size_t)win * 49 + 48) * 512 + n;
            tok[o] += acc[i][0] + bb;
        }
    }
}

// ---------------- window reverse -> NCHW ----------------
__global__ void k_reverse(const float* __restrict__ tok, float* __restrict__ out) {
    size_t gid = (size_t)blockIdx.x * 256 + threadIdx.x;
    if (gid >= (size_t)2 * 512 * 224 * 224) return;
    int xx = (int)(gid % 224);
    int y  = (int)((gid / 224) % 224);
    int c  = (int)((gid / (224 * 224)) % 512);
    int b  = (int)(gid / ((size_t)224 * 224 * 512));
    int win = (b * 32 + y / 7) * 32 + xx / 7;
    int t = (y % 7) * 7 + (xx % 7);
    out[gid] = tok[((size_t)win * 49 + t) * 512 + c];
}

extern "C" void kernel_launch(void* const* d_in, const int* in_sizes, int n_in,
                              void* d_out, int out_size, void* d_ws, size_t ws_size,
                              hipStream_t stream) {
    (void)in_sizes; (void)n_in; (void)out_size; (void)ws_size;
    const float* x        = (const float*)d_in[0];
    const float* W_pe     = (const float*)d_in[1];
    const float* b_pe     = (const float*)d_in[2];
    const float* W_pc     = (const float*)d_in[3];
    const float* b_pc     = (const float*)d_in[4];
    const float* ln1_g    = (const float*)d_in[5];
    const float* ln1_b    = (const float*)d_in[6];
    const float* W_qkv    = (const float*)d_in[7];
    const float* headsita = (const float*)d_in[8];
    const float* rpb      = (const float*)d_in[9];
    const float* W_thr    = (const float*)d_in[10];
    const float* W_out    = (const float*)d_in[11];
    const float* b_out    = (const float*)d_in[12];
    const float* ln2_g    = (const float*)d_in[13];
    const float* ln2_b    = (const float*)d_in[14];
    const float* W_ff1    = (const float*)d_in[15];
    const float* b_ff1    = (const float*)d_in[16];
    const float* W_ff2    = (const float*)d_in[17];
    const float* b_ff2    = (const float*)d_in[18];

    char* p = (char*)d_ws;
    float*    tok   = (float*)p;     p += (size_t)2048 * 49 * 512 * 4;
    float*    prob  = (float*)p;     p += (size_t)2048 * 49 * 4;
    _Float16* xnw   = (_Float16*)p;  p += (size_t)2048 * 64 * 512 * 2;
    _Float16* ow    = (_Float16*)p;  p += (size_t)2048 * 64 * 512 * 2;
    _Float16* WpeT  = (_Float16*)p;  p += (size_t)512 * 96 * 2;
    _Float16* WqkvT = (_Float16*)p;  p += (size_t)1536 * 512 * 2;
    _Float16* WoutT = (_Float16*)p;  p += (size_t)512 * 512 * 2;
    _Float16* Wff1T = (_Float16*)p;  p += (size_t)2048 * 512 * 2;
    _Float16* Wff2T = (_Float16*)p;  p += (size_t)512 * 2048 * 2;

    k_cvtT<<<(96 * 512 + 255) / 256, 256, 0, stream>>>(W_pe, WpeT, 96, 512);
    k_cvtT<<<(512 * 1536 + 255) / 256, 256, 0, stream>>>(W_qkv, WqkvT, 512, 1536);
    k_cvtT<<<(512 * 512 + 255) / 256, 256, 0, stream>>>(W_out, WoutT, 512, 512);
    k_cvtT<<<(512 * 2048 + 255) / 256, 256, 0, stream>>>(W_ff1, Wff1T, 512, 2048);
    k_cvtT<<<(2048 * 512 + 255) / 256, 256, 0, stream>>>(W_ff2, Wff2T, 2048, 512);

    k_embed<<<2048, 256, 0, stream>>>(x, W_pc, b_pc, WpeT, b_pe, tok, prob);
    for (int layer = 0; layer < 2; layer++) {
        k_ln<<<2048, 256, 0, stream>>>(tok, ln1_g, ln1_b, xnw);
        k_attn<<<16384, 128, 0, stream>>>(xnw, WqkvT, prob, rpb, headsita, W_thr, ow);
        k_proj<<<2048, 256, 0, stream>>>(ow, WoutT, b_out, tok);
        k_ln<<<2048, 256, 0, stream>>>(tok, ln2_g, ln2_b, xnw);
        k_ffn<<<2048, 256, 0, stream>>>(xnw, Wff1T, b_ff1, Wff2T, b_ff2, tok);
    }
    size_t total = (size_t)2 * 512 * 224 * 224;
    k_reverse<<<(unsigned)((total + 255) / 256), 256, 0, stream>>>(tok, (float*)d_out);
}
